// MultiHeadAttention_8134668058670
// MI455X (gfx1250) — matrix operand order repaired and checked by reading, not compiled, not run on hardware
//
#include <hip/hip_runtime.h>
#include <math.h>

// ---------------- constants (fixed by the reference) ----------------
#define N_HEAD 8
#define D_MODEL 512
#define D_K 64
#define D_V 64
#define B_SZ 4
#define SEQ 2048
#define LN_EPS 1e-5f
#define M_ROWS (B_SZ * SEQ)      // 8192
#define INV_TEMP 0.125f          // 1/sqrt(64)
#define SC_LD (SEQ + 16)         // LDS row pitch: +16 dwords kills bank aliasing
#define XR_LD (D_MODEL + 16)

// ---------------- WMMA types (gfx1250, wave32) ----------------
typedef __bf16 bf16_t;
typedef __attribute__((ext_vector_type(16))) __bf16 v16bf;
typedef __attribute__((ext_vector_type(8)))  __bf16 v8bf;
typedef __attribute__((ext_vector_type(8)))  float  v8f;

__device__ __forceinline__ v8f wmma_bf16(v16bf a, v16bf b, v8f c) {
  // (neg_a, A, neg_b, B, c_mod, C, reuse_a, reuse_b)
  return __builtin_amdgcn_wmma_f32_16x16x32_bf16(false, a, false, b, (short)0, c,
                                                 false, false);
}

// A-matrix 16x32 bf16 fragment (ISA 7.12.2): lane L<16 holds row M=L,
// K = {k0..k0+7, k0+16..k0+23}; lane L>=16 holds row M=L-16, K = {+8..+15, +24..+31}.
__device__ __forceinline__ v16bf load_a_f32(const float* row, int k0, int lane) {
  int klo = k0 + ((lane & 16) ? 8 : 0);
  v16bf a;
#pragma unroll
  for (int i = 0; i < 8; ++i) {
    a[i]     = (bf16_t)row[klo + i];
    a[i + 8] = (bf16_t)row[klo + 16 + i];
  }
  return a;
}
__device__ __forceinline__ v16bf load_a_bf16(const bf16_t* row, int k0, int lane) {
  int klo = k0 + ((lane & 16) ? 8 : 0);
  v8bf lo = *(const v8bf*)(row + klo);
  v8bf hi = *(const v8bf*)(row + klo + 16);
  v16bf a;
#pragma unroll
  for (int i = 0; i < 8; ++i) { a[i] = lo[i]; a[i + 8] = hi[i]; }
  return a;
}

// B-matrix 32x16 bf16 fragment: lane n<16 holds col N=n, K = {k0..k0+7, k0+16..k0+23};
// lane n>=16 holds col N=n-16, K = {+8..+15, +24..+31}. `row` = contiguous K-major
// data for this lane's column (B[k,n] contiguous in k).
__device__ __forceinline__ v16bf load_b_f32(const float* row, int k0, int lane) {
  int kk = k0 + ((lane & 16) ? 8 : 0);
  v16bf b;
#pragma unroll
  for (int i = 0; i < 8; ++i) { b[i] = (bf16_t)row[kk + i]; b[i + 8] = (bf16_t)row[kk + 16 + i]; }
  return b;
}
__device__ __forceinline__ v16bf load_b_bf16(const bf16_t* row, int k0, int lane) {
  int kk = k0 + ((lane & 16) ? 8 : 0);
  v8bf lo = *(const v8bf*)(row + kk);
  v8bf hi = *(const v8bf*)(row + kk + 16);
  v16bf b;
#pragma unroll
  for (int i = 0; i < 8; ++i) { b[i] = lo[i]; b[i + 8] = hi[i]; }
  return b;
}

// ================= Kernel 1: fused QKV projections =================
// grid (M/16, 512/16, 3), block 32 (1 wave). z selects q/k/v projection.
// Output q/k heads: bf16 [h][b][s][64]; v heads stored transposed [h][b][64][s].
__global__ void qkv_proj_kernel(const float* __restrict__ xq,
                                const float* __restrict__ xk,
                                const float* __restrict__ xv,
                                const float* __restrict__ w_qs, const float* __restrict__ b_qs,
                                const float* __restrict__ w_ks, const float* __restrict__ b_ks,
                                const float* __restrict__ w_vs, const float* __restrict__ b_vs,
                                bf16_t* __restrict__ qh, bf16_t* __restrict__ kh,
                                bf16_t* __restrict__ vh) {
  const int lane = threadIdx.x;
  const int m0 = blockIdx.x * 16;     // global row in [0, 8192)
  const int n0 = blockIdx.y * 16;     // output feature tile
  const int z = blockIdx.z;           // 0=q 1=k 2=v

  const float* X = (z == 0) ? xq : (z == 1) ? xk : xv;
  const float* W = (z == 0) ? w_qs : (z == 1) ? w_ks : w_vs;
  const float* Bi = (z == 0) ? b_qs : (z == 1) ? b_ks : b_vs;

  const float* arow = X + (size_t)(m0 + (lane & 15)) * D_MODEL;
  const float* brow = W + (size_t)(n0 + (lane & 15)) * D_MODEL; // W[f][d], B[k=d,n=f]

  // two independent accumulator chains -> no back-to-back WMMA C-dependency
  v8f acc0 = {}, acc1 = {};
#pragma unroll 4
  for (int k0 = 0; k0 < D_MODEL; k0 += 64) {
    acc0 = wmma_bf16(load_a_f32(arow, k0, lane), load_b_f32(brow, k0, lane), acc0);
    acc1 = wmma_bf16(load_a_f32(arow, k0 + 32, lane), load_b_f32(brow, k0 + 32, lane), acc1);
  }

  const int n = lane & 15;
  const int f = n0 + n;
  const int h = f >> 6, dk = f & 63;
  const int bidx = m0 >> 11;                 // row / SEQ
  const int mbase = (lane & 16) ? 8 : 0;
  const float bias = Bi[f];

  if (z == 2) {
    // vh transposed: [h][b][dv][s], lane writes 8 contiguous bf16 along s
    v8bf pack;
#pragma unroll
    for (int r = 0; r < 8; ++r) pack[r] = (bf16_t)(acc0[r] + acc1[r] + bias);
    size_t base = (((size_t)h * B_SZ + bidx) * D_V + dk) * SEQ + (size_t)((m0 & (SEQ - 1)) + mbase);
    *(v8bf*)(vh + base) = pack;
  } else {
    bf16_t* dst = (z == 0) ? qh : kh;        // [h][b][s][64]
#pragma unroll
    for (int r = 0; r < 8; ++r) {
      int s = (m0 & (SEQ - 1)) + mbase + r;
      dst[(((size_t)h * B_SZ + bidx) * SEQ + s) * D_K + dk] = (bf16_t)(acc0[r] + acc1[r] + bias);
    }
  }
}

// ================= Kernel 2: scores + softmax + P.V =================
// grid (SEQ/16, B, H), block 256 (8 waves).
// Phase1: score strip 16 x 2048 -> LDS. Phase2: fp32 softmax, write attn to
// d_out (only HBM write of the 537MB matrix). Phase3: all 8 waves do P.V
// (dv-tile x K-half split) with an LDS partial reduction.
__global__ __launch_bounds__(256) void attn_kernel(const bf16_t* __restrict__ qh,
                                                   const bf16_t* __restrict__ kh,
                                                   const bf16_t* __restrict__ vh,
                                                   float* __restrict__ attn_out,
                                                   float* __restrict__ ctx) {
  __shared__ float sc[16][SC_LD];    // ~129 KB padded score strip
  __shared__ float red[16][16];
  __shared__ float rowstat[16];
  __shared__ float part[4][16][17];  // P.V partial tiles (padded)

  const int tid = threadIdx.x;
  const int wave = tid >> 5;
  const int lane = tid & 31;
  const int q0 = blockIdx.x * 16;
  const int b = blockIdx.y;
  const int h = blockIdx.z;

  const bf16_t* Q = qh + ((size_t)h * B_SZ + b) * SEQ * D_K;
  const bf16_t* K = kh + ((size_t)h * B_SZ + b) * SEQ * D_K;
  const bf16_t* V = vh + ((size_t)h * B_SZ + b) * D_V * SEQ; // [64][2048]

  // ---- phase 1: S = Q.K^T * invTemp (2 independent WMMAs per 16-wide k-tile) ----
  {
    const bf16_t* arow = Q + (size_t)(q0 + (lane & 15)) * D_K;
    v16bf a0 = load_a_bf16(arow, 0, lane);
    v16bf a1 = load_a_bf16(arow, 32, lane);
    const int n = lane & 15;
    const int mbase = (lane & 16) ? 8 : 0;
#pragma unroll 2
    for (int kt = wave; kt < SEQ / 16; kt += 8) {
      const int kcol0 = kt * 16;
      const bf16_t* brow = K + (size_t)(kcol0 + (lane & 15)) * D_K;
      // prefetch this lane's row for the next k-tile handled by this wave
      __builtin_prefetch((const void*)(brow + (size_t)8 * 16 * D_K), 0, 1);
      v8f acc0 = {}, acc1 = {};
      acc0 = wmma_bf16(a0, load_b_bf16(brow, 0, lane), acc0);
      acc1 = wmma_bf16(a1, load_b_bf16(brow, 32, lane), acc1);
#pragma unroll
      for (int r = 0; r < 8; ++r)
        sc[mbase + r][kcol0 + n] = (acc0[r] + acc1[r]) * INV_TEMP;
    }
  }
  __syncthreads();

  // ---- phase 2: fp32 softmax over each of the 16 rows ----
  {
    const int row = tid >> 4, sub = tid & 15;
    float mx = -1e30f;
    for (int j = sub; j < SEQ; j += 16) mx = fmaxf(mx, sc[row][j]);
    red[row][sub] = mx;
    __syncthreads();
    if (sub == 0) {
      float m2 = red[row][0];
#pragma unroll
      for (int i = 1; i < 16; ++i) m2 = fmaxf(m2, red[row][i]);
      rowstat[row] = m2;
    }
    __syncthreads();
    const float rm = rowstat[row];
    float sum = 0.f;
    for (int j = sub; j < SEQ; j += 16) {
      float e = __expf(sc[row][j] - rm);
      sc[row][j] = e;
      sum += e;
    }
    red[row][sub] = sum;
    __syncthreads();
    if (sub == 0) {
      float t = 0.f;
#pragma unroll
      for (int i = 0; i < 16; ++i) t += red[row][i];
      rowstat[row] = 1.f / t;
    }
    __syncthreads();
    const float inv = rowstat[row];
    float* arow_out = attn_out + (((size_t)h * B_SZ + b) * SEQ + (q0 + row)) * SEQ;
    for (int j = sub; j < SEQ; j += 16) {
      float p = sc[row][j] * inv;
      sc[row][j] = p;
      arow_out[j] = p;           // the only HBM write of attn
    }
  }
  __syncthreads();

  // ---- phase 3: ctx = P.V, all 8 waves: wave = (khalf<<2) | dvtile ----
  {
    const int dv0 = (wave & 3) * 16;
    const int kbase = (wave >> 2) * (SEQ / 2);
    const int mrow = lane & 15;
    const int klo_sel = (lane & 16) ? 8 : 0;
    const bf16_t* brow = V + (size_t)(dv0 + (lane & 15)) * SEQ; // contiguous in k
    v8f acc = {};
    for (int k0 = kbase; k0 < kbase + SEQ / 2; k0 += 32) {
      v16bf a;
      const int klo = k0 + klo_sel;
#pragma unroll
      for (int i = 0; i < 8; ++i) {
        a[i]     = (bf16_t)sc[mrow][klo + i];
        a[i + 8] = (bf16_t)sc[mrow][klo + 16 + i];
      }
      acc = wmma_bf16(a, load_b_bf16(brow, k0, lane), acc);
    }
    const int n = lane & 15;
    const int mbase = (lane & 16) ? 8 : 0;
    if (wave >= 4) {
#pragma unroll
      for (int r = 0; r < 8; ++r) part[wave - 4][mbase + r][n] = acc[r];
    }
    __syncthreads();
    if (wave < 4) {
#pragma unroll
      for (int r = 0; r < 8; ++r) {
        size_t idx = ((size_t)b * SEQ + q0 + mbase + r) * (N_HEAD * D_V) + h * D_V + dv0 + n;
        ctx[idx] = acc[r] + part[wave][mbase + r][n];
      }
    }
  }
}

// ================= Kernel 3: FC + residual + LayerNorm =================
// grid M/16, block 256 (8 waves, 4 n-tiles each over 512 features).
__global__ __launch_bounds__(256) void fc_ln_kernel(const float* __restrict__ ctx,
                                                    const float* __restrict__ w_fc,
                                                    const float* __restrict__ b_fc,
                                                    const float* __restrict__ resid,
                                                    const float* __restrict__ ln_g,
                                                    const float* __restrict__ ln_b,
                                                    float* __restrict__ xout) {
  __shared__ float xr[16][XR_LD];    // padded row buffer
  __shared__ float red[16][16];
  __shared__ float smu[16];
  __shared__ float sinv[16];

  const int tid = threadIdx.x;
  const int wave = tid >> 5;
  const int lane = tid & 31;
  const int m0 = blockIdx.x * 16;

  // ---- GEMM: Y = ctx . W_fc^T  (B[k=f,n=d] = w_fc[d][f], contiguous in f) ----
  {
    const float* arow = ctx + (size_t)(m0 + (lane & 15)) * D_MODEL;
    const float* brow[4];
    int nt0[4];
#pragma unroll
    for (int t = 0; t < 4; ++t) {
      nt0[t] = (wave * 4 + t) * 16;
      brow[t] = w_fc + (size_t)(nt0[t] + (lane & 15)) * D_MODEL;
    }
    v8f acc[4] = {{}, {}, {}, {}};
    for (int k0 = 0; k0 < D_MODEL; k0 += 32) {
      v16bf a = load_a_f32(arow, k0, lane);
#pragma unroll
      for (int t = 0; t < 4; ++t)
        acc[t] = wmma_bf16(a, load_b_f32(brow[t], k0, lane), acc[t]);
    }
    const int n = lane & 15;
    const int mbase = (lane & 16) ? 8 : 0;
#pragma unroll
    for (int t = 0; t < 4; ++t)
#pragma unroll
      for (int r = 0; r < 8; ++r) xr[mbase + r][nt0[t] + n] = acc[t][r];
  }
  __syncthreads();

  // ---- bias + residual + LayerNorm ----
  const int row = tid >> 4, sub = tid & 15;
  const float* res = resid + (size_t)(m0 + row) * D_MODEL;
  float s = 0.f;
  for (int j = sub; j < D_MODEL; j += 16) {
    float v = xr[row][j] + b_fc[j] + res[j];
    xr[row][j] = v;
    s += v;
  }
  red[row][sub] = s;
  __syncthreads();
  if (sub == 0) {
    float t = 0.f;
#pragma unroll
    for (int i = 0; i < 16; ++i) t += red[row][i];
    smu[row] = t * (1.f / D_MODEL);
  }
  __syncthreads();
  const float mu = smu[row];
  float vs = 0.f;
  for (int j = sub; j < D_MODEL; j += 16) {
    float d = xr[row][j] - mu;
    vs += d * d;
  }
  red[row][sub] = vs;
  __syncthreads();
  if (sub == 0) {
    float t = 0.f;
#pragma unroll
    for (int i = 0; i < 16; ++i) t += red[row][i];
    sinv[row] = rsqrtf(t * (1.f / D_MODEL) + LN_EPS);
  }
  __syncthreads();
  const float inv = sinv[row];
  float* orow = xout + (size_t)(m0 + row) * D_MODEL;
  for (int j = sub; j < D_MODEL; j += 16)
    orow[j] = (xr[row][j] - mu) * inv * ln_g[j] + ln_b[j];
}

// ================= host launcher =================
extern "C" void kernel_launch(void* const* d_in, const int* in_sizes, int n_in,
                              void* d_out, int out_size, void* d_ws, size_t ws_size,
                              hipStream_t stream) {
  (void)in_sizes; (void)n_in; (void)out_size; (void)ws_size;
  const float* q    = (const float*)d_in[0];
  const float* k    = (const float*)d_in[1];
  const float* v    = (const float*)d_in[2];
  const float* w_qs = (const float*)d_in[3];
  const float* b_qs = (const float*)d_in[4];
  const float* w_ks = (const float*)d_in[5];
  const float* b_ks = (const float*)d_in[6];
  const float* w_vs = (const float*)d_in[7];
  const float* b_vs = (const float*)d_in[8];
  const float* w_fc = (const float*)d_in[9];
  const float* b_fc = (const float*)d_in[10];
  const float* ln_g = (const float*)d_in[11];
  const float* ln_b = (const float*)d_in[12];

  char* ws = (char*)d_ws;
  bf16_t* qh = (bf16_t*)(ws);                          // 8 MB
  bf16_t* kh = (bf16_t*)(ws + ((size_t)8 << 20));      // 8 MB
  bf16_t* vh = (bf16_t*)(ws + ((size_t)16 << 20));     // 8 MB (transposed)
  float*  ctx = (float*)(ws + ((size_t)24 << 20));     // 16 MB

  float* xout = (float*)d_out;                                   // [4,2048,512]
  float* attn = (float*)d_out + (size_t)B_SZ * SEQ * D_MODEL;    // [32,2048,2048]

  qkv_proj_kernel<<<dim3(M_ROWS / 16, D_MODEL / 16, 3), 32, 0, stream>>>(
      q, k, v, w_qs, b_qs, w_ks, b_ks, w_vs, b_vs, qh, kh, vh);
  attn_kernel<<<dim3(SEQ / 16, B_SZ, N_HEAD), 256, 0, stream>>>(
      qh, kh, vh, attn, ctx);
  fc_ln_kernel<<<dim3(M_ROWS / 16), 256, 0, stream>>>(
      ctx, w_fc, b_fc, q, ln_g, ln_b, xout);
}
